// dLSTM_884763263441
// MI455X (gfx1250) — matrix operand order repaired and checked by reading, not compiled, not run on hardware
//
#include <hip/hip_runtime.h>
#include <math.h>

// Problem constants (fixed by the reference).
#define L_   2
#define H_   2048
#define OUT_ 1024
#define B_   64
#define T_   256
#define KC_  4096   // concatenated K = 2*H (x | h)

typedef __attribute__((ext_vector_type(16))) __bf16 v16bf;
typedef __attribute__((ext_vector_type(8)))  __bf16 v8bf;
typedef __attribute__((ext_vector_type(8)))  float  v8f;

__device__ __forceinline__ __bf16 f2bf(float f) {
  union { float f; unsigned u; } v; v.f = f;
  unsigned r = v.u + 0x7FFFu + ((v.u >> 16) & 1u);   // round-to-nearest-even
  unsigned short h = (unsigned short)(r >> 16);
  __bf16 o; __builtin_memcpy(&o, &h, 2); return o;
}

__device__ __forceinline__ float sigmoidf_(float x) {
  return 1.0f / (1.0f + __expf(-x));
}

// ---------------------------------------------------------------------------
// One-time weight pack: fp32 -> bf16, W_ih|W_hh concatenated along K so the
// per-step kernel's B-fragment loads are contiguous b128s.
// Wcat layout: [L][8192][4096] row-major bf16; WoutB: [1024][2048] bf16.
// ---------------------------------------------------------------------------
__global__ void pack_weights_kernel(const float* __restrict__ Wih,
                                    const float* __restrict__ Whh,
                                    const float* __restrict__ WoutF,
                                    __bf16* __restrict__ Wcat,
                                    __bf16* __restrict__ WoutB) {
  const long total1 = (long)L_ * 4 * H_ * KC_;   // 67,108,864
  const long total2 = (long)OUT_ * H_;           //  2,097,152
  const long stride = (long)gridDim.x * blockDim.x;
  for (long idx = (long)blockIdx.x * blockDim.x + threadIdx.x;
       idx < total1 + total2; idx += stride) {
    if (idx < total1) {
      const long l   = idx / ((long)4 * H_ * KC_);
      const long rem = idx % ((long)4 * H_ * KC_);
      const long r   = rem / KC_;
      const long k   = rem % KC_;
      const float v  = (k < H_)
          ? Wih[(l * 4 * H_ + r) * H_ + k]
          : Whh[(l * 4 * H_ + r) * H_ + (k - H_)];
      Wcat[idx] = f2bf(v);
    } else {
      const long j = idx - total1;
      WoutB[j] = f2bf(WoutF[j]);
    }
  }
}

// ---------------------------------------------------------------------------
// State init: Xcat[l][0][:, 0:2048] = x0 (zeros for layer 0),
//             Xcat[l][0][:, 2048:4096] = bf16(h0[l]),  Cst[l] = c0[l].
// Xcat layout: [L][2 ping-pong][64][4096] bf16.
// ---------------------------------------------------------------------------
__global__ void init_state_kernel(const float* __restrict__ h0,
                                  const float* __restrict__ c0,
                                  __bf16* __restrict__ Xcat,
                                  float* __restrict__ Cst) {
  const int stride = gridDim.x * blockDim.x;
  for (int idx = blockIdx.x * blockDim.x + threadIdx.x;
       idx < B_ * H_; idx += stride) {
    const int row = idx / H_;
    const int col = idx % H_;
    // layer-0 input at t=0 is zeros
    Xcat[((0 * 2 + 0) * B_ + row) * (long)KC_ + col] = f2bf(0.0f);
    #pragma unroll
    for (int l = 0; l < L_; ++l) {
      Xcat[((l * 2 + 0) * B_ + row) * (long)KC_ + H_ + col] =
          f2bf(h0[((long)l * B_ + row) * H_ + col]);
      Cst[((long)l * B_ + row) * H_ + col] = c0[((long)l * B_ + row) * H_ + col];
    }
  }
}

// ---------------------------------------------------------------------------
// Fused LSTM step for one layer.
//   gates[64,8192] = A[64,4096] @ W^T   (W row-major [8192,4096] bf16)
// Grid: 128 WGs (16 hidden cols each) x 128 threads (4 waves, K split 4-way).
// Wave-private accumulators: 4 M-tiles x 4 gates x v8f. LDS cross-wave reduce,
// then wave 0 applies the LSTM cell and scatters h (bf16) / c (fp32).
// ---------------------------------------------------------------------------
__global__ __launch_bounds__(128) void lstm_step_kernel(
    const __bf16* __restrict__ A,     // [64][4096]
    const __bf16* __restrict__ W,     // [8192][4096]
    const float*  __restrict__ bih,   // [8192]
    const float*  __restrict__ bhh,   // [8192]
    float*        __restrict__ Cst,   // [64][2048], updated in place
    __bf16*       __restrict__ XhDst, // [64][4096]: h written at col 2048+
    __bf16*       __restrict__ XiDst) // [64][4096]: h written at col 0 (next consumer's input)
{
  __shared__ float red[3][4][4][32][8];   // 48 KB: partials of waves 1..3

  const int wg    = blockIdx.x;           // hidden tile 0..127
  const int lane  = threadIdx.x & 31;
  const int wave  = threadIdx.x >> 5;     // K-split index 0..3
  const int col16 = lane & 15;
  const int n0    = wg * 16;
  const int aoff  = (lane >> 4) * 8;      // A frag: lanes16-31 hold K+8 chunks
  const int boff  = (lane >> 4) * 8;      // B frag: lanes16-31 hold K+8 chunks

  v8f acc[4][4];
  #pragma unroll
  for (int m = 0; m < 4; ++m)
    #pragma unroll
    for (int g = 0; g < 4; ++g) {
      v8f z = {0.f, 0.f, 0.f, 0.f, 0.f, 0.f, 0.f, 0.f};
      acc[m][g] = z;
    }

  const int kbase = wave * (KC_ / 4);     // 1024 K per wave

  #pragma unroll 1
  for (int kk = 0; kk < (KC_ / 4) / 32; ++kk) {   // 32 iterations
    const int k0 = kbase + kk * 32;

    // A fragments: 4 M-tiles, two contiguous b128 loads each
    v16bf afr[4];
    #pragma unroll
    for (int m = 0; m < 4; ++m) {
      const __bf16* ap = A + (long)(m * 16 + col16) * KC_ + k0 + aoff;
      v16bf t;
      *((v8bf*)&t)     = *(const v8bf*)ap;          // K = k0+aoff .. +7
      *((v8bf*)&t + 1) = *(const v8bf*)(ap + 16);   // K = k0+16+aoff .. +7
      afr[m] = t;
    }

    // For each gate: one B fragment, reused across 4 M-tiles
    #pragma unroll
    for (int g = 0; g < 4; ++g) {
      const __bf16* bp = W + (long)(g * H_ + n0 + col16) * KC_ + k0 + boff;
      v16bf bfr;
      *((v8bf*)&bfr)     = *(const v8bf*)bp;        // K = k0+boff .. +7
      *((v8bf*)&bfr + 1) = *(const v8bf*)(bp + 16); // K = k0+16+boff .. +7
      __builtin_prefetch(bp + 32, 0, 1);            // next k-iter of this row
      #pragma unroll
      for (int m = 0; m < 4; ++m) {
        acc[m][g] = __builtin_amdgcn_wmma_f32_16x16x32_bf16(
            false, afr[m], false, bfr, (short)0, acc[m][g], false, false);
      }
    }
  }

  // Cross-wave K reduction through LDS
  if (wave > 0) {
    #pragma unroll
    for (int m = 0; m < 4; ++m)
      #pragma unroll
      for (int g = 0; g < 4; ++g)
        *(v8f*)&red[wave - 1][m][g][lane][0] = acc[m][g];
  }
  __syncthreads();

  if (wave == 0) {
    #pragma unroll
    for (int w = 0; w < 3; ++w)
      #pragma unroll
      for (int m = 0; m < 4; ++m)
        #pragma unroll
        for (int g = 0; g < 4; ++g)
          acc[m][g] += *(v8f*)&red[w][m][g][lane][0];

    // LSTM cell (gate order i,f,g,o as in the reference)
    const int col = n0 + col16;
    const float bi = bih[0 * H_ + col] + bhh[0 * H_ + col];
    const float bf = bih[1 * H_ + col] + bhh[1 * H_ + col];
    const float bg = bih[2 * H_ + col] + bhh[2 * H_ + col];
    const float bo = bih[3 * H_ + col] + bhh[3 * H_ + col];

    #pragma unroll
    for (int m = 0; m < 4; ++m) {
      #pragma unroll
      for (int v = 0; v < 8; ++v) {
        const int row = m * 16 + v + ((lane >> 4) << 3);  // C/D layout: lanes16-31 -> M+8
        const float gi = acc[m][0][v] + bi;
        const float gf = acc[m][1][v] + bf;
        const float gg = acc[m][2][v] + bg;
        const float go = acc[m][3][v] + bo;
        const float co = Cst[(long)row * H_ + col];
        const float cn = sigmoidf_(gf) * co + sigmoidf_(gi) * tanhf(gg);
        const float hn = sigmoidf_(go) * tanhf(cn);
        Cst[(long)row * H_ + col] = cn;
        const __bf16 hb = f2bf(hn);
        XhDst[(long)row * KC_ + H_ + col] = hb;   // recurrent h for next step
        XiDst[(long)row * KC_ + col]      = hb;   // input for next consumer
      }
    }
  }
}

// ---------------------------------------------------------------------------
// Output projection: y[64,1024] = Htop[64,2048] @ Wout^T + b_out   (fp32 out)
// Grid: 64 WGs x 128 threads (4 waves, K split 4-way over K=2048).
// ---------------------------------------------------------------------------
__global__ __launch_bounds__(128) void proj_step_kernel(
    const __bf16* __restrict__ Htop,  // [64][4096] buffer, cols 0..2047 valid
    const __bf16* __restrict__ W,     // [1024][2048] bf16 row-major
    const float*  __restrict__ bout,  // [1024]
    float*        __restrict__ Y)     // [64][1024]
{
  __shared__ float red[3][4][32][8];  // 12 KB

  const int wg    = blockIdx.x;       // 0..63
  const int lane  = threadIdx.x & 31;
  const int wave  = threadIdx.x >> 5;
  const int col16 = lane & 15;
  const int n0    = wg * 16;
  const int aoff  = (lane >> 4) * 8;
  const int boff  = (lane >> 4) * 8;

  v8f acc[4];
  #pragma unroll
  for (int m = 0; m < 4; ++m) {
    v8f z = {0.f, 0.f, 0.f, 0.f, 0.f, 0.f, 0.f, 0.f};
    acc[m] = z;
  }

  const int kbase = wave * (H_ / 4);  // 512 K per wave

  #pragma unroll 1
  for (int kk = 0; kk < (H_ / 4) / 32; ++kk) {   // 16 iterations
    const int k0 = kbase + kk * 32;

    v16bf afr[4];
    #pragma unroll
    for (int m = 0; m < 4; ++m) {
      const __bf16* ap = Htop + (long)(m * 16 + col16) * KC_ + k0 + aoff;
      v16bf t;
      *((v8bf*)&t)     = *(const v8bf*)ap;
      *((v8bf*)&t + 1) = *(const v8bf*)(ap + 16);
      afr[m] = t;
    }

    const __bf16* bp = W + (long)(n0 + col16) * H_ + k0 + boff;
    v16bf bfr;
    *((v8bf*)&bfr)     = *(const v8bf*)bp;
    *((v8bf*)&bfr + 1) = *(const v8bf*)(bp + 16);
    __builtin_prefetch(bp + 32, 0, 1);

    #pragma unroll
    for (int m = 0; m < 4; ++m) {
      acc[m] = __builtin_amdgcn_wmma_f32_16x16x32_bf16(
          false, afr[m], false, bfr, (short)0, acc[m], false, false);
    }
  }

  if (wave > 0) {
    #pragma unroll
    for (int m = 0; m < 4; ++m)
      *(v8f*)&red[wave - 1][m][lane][0] = acc[m];
  }
  __syncthreads();

  if (wave == 0) {
    #pragma unroll
    for (int w = 0; w < 3; ++w)
      #pragma unroll
      for (int m = 0; m < 4; ++m)
        acc[m] += *(v8f*)&red[w][m][lane][0];

    const int col = n0 + col16;
    const float bb = bout[col];
    #pragma unroll
    for (int m = 0; m < 4; ++m) {
      #pragma unroll
      for (int v = 0; v < 8; ++v) {
        const int row = m * 16 + v + ((lane >> 4) << 3);
        Y[(long)row * OUT_ + col] = acc[m][v] + bb;
      }
    }
  }
}

// ---------------------------------------------------------------------------
extern "C" void kernel_launch(void* const* d_in, const int* in_sizes, int n_in,
                              void* d_out, int out_size, void* d_ws, size_t ws_size,
                              hipStream_t stream) {
  const float* h0    = (const float*)d_in[0];
  const float* c0    = (const float*)d_in[1];
  const float* Wih   = (const float*)d_in[2];
  const float* Whh   = (const float*)d_in[3];
  const float* bih   = (const float*)d_in[4];
  const float* bhh   = (const float*)d_in[5];
  const float* WoutF = (const float*)d_in[6];
  const float* bout  = (const float*)d_in[7];
  float* Y = (float*)d_out;

  // Workspace layout (all bf16 weights L2-resident: ~138 MB < 192 MB L2)
  char* ws = (char*)d_ws;
  const size_t WCAT_BYTES = (size_t)L_ * 4 * H_ * KC_ * 2;   // 134,217,728
  const size_t WOUT_BYTES = (size_t)OUT_ * H_ * 2;           //   4,194,304
  const size_t XCAT_BYTES = (size_t)L_ * 2 * B_ * KC_ * 2;   //   2,097,152
  __bf16* Wcat  = (__bf16*)ws;
  __bf16* WoutB = (__bf16*)(ws + WCAT_BYTES);
  __bf16* Xcat  = (__bf16*)(ws + WCAT_BYTES + WOUT_BYTES);
  float*  Cst   = (float*)(ws + WCAT_BYTES + WOUT_BYTES + XCAT_BYTES);

  pack_weights_kernel<<<4096, 256, 0, stream>>>(Wih, Whh, WoutF, Wcat, WoutB);
  init_state_kernel<<<256, 256, 0, stream>>>(h0, c0, Xcat, Cst);

  auto Xc = [&](int l, int p) -> __bf16* {
    return Xcat + ((size_t)(l * 2 + p)) * B_ * KC_;
  };

  for (int t = 0; t < T_; ++t) {
    const int p = t & 1;
    // Layer 0: reads Xcat[0][p]; h -> Xcat[0][1-p] (recurrent), Xcat[1][p] (layer-1 input)
    lstm_step_kernel<<<H_ / 16, 128, 0, stream>>>(
        Xc(0, p), Wcat, bih, bhh, Cst, Xc(0, 1 - p), Xc(1, p));
    // Layer 1: reads Xcat[1][p]; h -> Xcat[1][1-p] (recurrent), Xcat[0][1-p] (next-step input)
    lstm_step_kernel<<<H_ / 16, 128, 0, stream>>>(
        Xc(1, p), Wcat + (size_t)4 * H_ * KC_, bih + 4 * H_, bhh + 4 * H_,
        Cst + (size_t)B_ * H_, Xc(1, 1 - p), Xc(0, 1 - p));
    // y_t from the top-layer hidden just written to Xcat[0][1-p][:, 0:2048]
    proj_step_kernel<<<OUT_ / 16, 128, 0, stream>>>(
        Xc(0, 1 - p), WoutB, bout, Y + (size_t)t * B_ * OUT_);
  }
}
